// DeepseekV2MoE_14190571946209
// MI455X (gfx1250) — matrix operand order repaired and checked by reading, not compiled, not run on hardware
//
#include <hip/hip_runtime.h>
#include <hip/hip_bf16.h>

// ---------------- problem constants ----------------
#define TT   512      // tokens
#define HH   2048     // hidden
#define EE   256      // experts
#define GG   8        // groups
#define EPG  32       // experts per group
#define TOPK 6
#define II   1408     // moe intermediate
#define RR   128      // tucker ranks r1 = r2 = 128
#define PAIRS (TT*TOPK)          // 3072
#define MAXT  432                // worst-case 16-row tiles: 256 + (3072-256)/16
#define MAXS  (MAXT*16)          // 6912 padded slots
#define NT    4                  // n-tiles per wave (16x64 strip); all N % 64 == 0

typedef __bf16 bf16_t;
typedef bf16_t v16bf __attribute__((ext_vector_type(16)));
typedef bf16_t v8bf  __attribute__((ext_vector_type(8)));
typedef bf16_t v4bf  __attribute__((ext_vector_type(4)));
typedef float  v8f   __attribute__((ext_vector_type(8)));

// ---------------- fp32 -> bf16 convert (vectorized x4) ----------------
__global__ void cvt_bf16_kernel(const float* __restrict__ in, bf16_t* __restrict__ out, long n4) {
    long i = (long)blockIdx.x * 256 + threadIdx.x;
    if (i >= n4) return;
    float4 v = ((const float4*)in)[i];
    v4bf o;
    o[0] = (bf16_t)v.x; o[1] = (bf16_t)v.y; o[2] = (bf16_t)v.z; o[3] = (bf16_t)v.w;
    ((v4bf*)out)[i] = o;
}

// ---------------- fp32 [z][rows][cols] -> bf16 [z][cols][rows] ----------------
__global__ __launch_bounds__(256)
void transpose_bf16_kernel(const float* __restrict__ in, bf16_t* __restrict__ out,
                           int rows, int cols) {
    __shared__ bf16_t tile[32][33];
    long zoff = (long)blockIdx.z * rows * cols;
    in  += zoff; out += zoff;
    int c0 = blockIdx.x * 32, r0 = blockIdx.y * 32;
    int tx = threadIdx.x & 31, ty = threadIdx.x >> 5;    // 32 x 8
    #pragma unroll
    for (int i = ty; i < 32; i += 8) {
        int r = r0 + i, c = c0 + tx;
        tile[i][tx] = (r < rows && c < cols) ? (bf16_t)in[(long)r * cols + c] : (bf16_t)0.0f;
    }
    __syncthreads();
    #pragma unroll
    for (int i = ty; i < 32; i += 8) {
        int c = c0 + i, r = r0 + tx;
        if (c < cols && r < rows) out[(long)c * rows + r] = tile[tx][i];
    }
}

// ---------------- fp32 router GEMM (exact routing) ----------------
__global__ __launch_bounds__(256)
void router_kernel(const float* __restrict__ x, const float* __restrict__ w,
                   float* __restrict__ logits) {
    __shared__ float xs[HH];
    int t = blockIdx.x;
    for (int i = threadIdx.x; i < HH; i += 256) xs[i] = x[(size_t)t * HH + i];
    __syncthreads();
    int e = threadIdx.x;            // 256 experts == 256 threads
    float acc = 0.f;
    #pragma unroll 8
    for (int k = 0; k < HH; ++k) acc = fmaf(xs[k], w[(size_t)k * EE + e], acc);
    logits[(size_t)t * EE + e] = acc;
}

// ---------------- top-6 + softmax-over-selected (1 wave32 / token) ----------------
__global__ __launch_bounds__(32)
void topk_kernel(const float* __restrict__ logits, int* __restrict__ selected,
                 float* __restrict__ weights) {
    int t = blockIdx.x, lane = threadIdx.x;
    float l[8]; bool taken[8];
    #pragma unroll
    for (int i = 0; i < 8; ++i) { l[i] = logits[(size_t)t * EE + i * 32 + lane]; taken[i] = false; }
    float topl[TOPK]; int tope[TOPK];
    for (int kk = 0; kk < TOPK; ++kk) {
        float best = -3.0e38f; int bi = -1;
        #pragma unroll
        for (int i = 0; i < 8; ++i)
            if (!taken[i] && l[i] > best) { best = l[i]; bi = i; }
        int be = (bi >= 0) ? (bi * 32 + lane) : 0x7fffffff;
        for (int off = 16; off; off >>= 1) {   // lowest-index tie break, like lax.top_k
            float ob = __shfl_xor(best, off, 32);
            int   oe = __shfl_xor(be,   off, 32);
            if (ob > best || (ob == best && oe < be)) { best = ob; be = oe; }
        }
        topl[kk] = best; tope[kk] = be;
        if ((be & 31) == lane) taken[be >> 5] = true;
    }
    if (lane == 0) {
        float m = topl[0], s = 0.f, w6[TOPK];
        #pragma unroll
        for (int k = 0; k < TOPK; ++k) { w6[k] = __expf(topl[k] - m); s += w6[k]; }
        float inv = 1.f / s;
        #pragma unroll
        for (int k = 0; k < TOPK; ++k) {
            weights[t * TOPK + k]  = w6[k] * inv;
            selected[t * TOPK + k] = tope[k];
        }
    }
}

// ---------------- routing bookkeeping ----------------
__global__ void zero_int_kernel(int* a, int n) {
    int i = blockIdx.x * 256 + threadIdx.x;
    if (i < n) a[i] = 0;
}

__global__ void count_kernel(const int* __restrict__ selected, int* __restrict__ counts) {
    int p = blockIdx.x * 256 + threadIdx.x;
    if (p < PAIRS) atomicAdd(&counts[selected[p]], 1);
}

__global__ __launch_bounds__(256)
void plan_kernel(const int* __restrict__ counts, int* __restrict__ padded_start,
                 int* __restrict__ tile_start, int* __restrict__ tile_expert,
                 int* __restrict__ sorted_pair, int* __restrict__ rowGH) {
    for (int i = threadIdx.x; i < MAXT; i += 256) tile_expert[i] = -1;
    for (int i = threadIdx.x; i < MAXS; i += 256) { sorted_pair[i] = -1; rowGH[i] = -1; }
    __syncthreads();
    if (threadIdx.x == 0) {
        int off = 0, toff = 0;
        for (int e = 0; e < EE; ++e) {
            padded_start[e] = off; tile_start[e] = toff;
            int ct = (counts[e] + 15) >> 4;
            off += ct << 4; toff += ct;
        }
    }
    __syncthreads();
    int e = threadIdx.x;                       // exactly 256 experts
    int ct = (counts[e] + 15) >> 4;
    int ts = tile_start[e];
    for (int i = 0; i < ct; ++i) tile_expert[ts + i] = e;
}

__global__ void scatter_kernel(const int* __restrict__ selected,
                               const int* __restrict__ padded_start,
                               int* __restrict__ cursor,
                               int* __restrict__ sorted_pair, int* __restrict__ rowGH) {
    int p = blockIdx.x * 256 + threadIdx.x;
    if (p >= PAIRS) return;
    int e = selected[p];
    int g = e >> 5;                  // e / EPG
    int t = p / TOPK;
    int pos  = atomicAdd(&cursor[e], 1);
    int slot = padded_start[e] + pos;
    sorted_pair[slot] = p;
    rowGH[slot]       = g * TT + t;  // row into [G,T,R] H1 buffers
}

// ---------------- bf16 WMMA GEMM, 16x64 strip per wave, 2-stage pipeline ----
// All operands bf16. B layout is [N][K] (K contiguous) so every fragment is a
// 32-byte contiguous per-lane load. N % 64 == 0 and K % 64 == 0 required.
//  A: bf16 row-major, row stride lda; row gather a_rows (-1 => zrow dummy zeros)
//  B base offset (elements): z*b_z + e*b_e + g*b_g  (e,g from tile_expert when given)
//  C: bf16 (Cb) or fp32 (Cf); row stride ldc; optional row scatter (c_rows, -1 => skip)
__global__ __launch_bounds__(128)
void wmma_gemm(const bf16_t* __restrict__ A, int lda, long a_z, const int* __restrict__ a_rows,
               const bf16_t* __restrict__ B, long b_z, long b_e, long b_g,
               bf16_t* __restrict__ Cb, float* __restrict__ Cf, int ldc, long c_z,
               const int* __restrict__ c_rows,
               int M, int N, int K, const int* __restrict__ tile_expert,
               const bf16_t* __restrict__ zrow) {
    int wave = threadIdx.x >> 5;
    int lane = threadIdx.x & 31;
    int m_tile  = blockIdx.x;
    int n_tile0 = (blockIdx.y * 4 + wave) * NT;
    int z = blockIdx.z;

    int e = 0, g = 0;
    if (tile_expert) {
        e = tile_expert[m_tile];
        if (e < 0) return;           // inactive padded tile (uniform per block)
        g = e >> 5;
    }
    if (m_tile * 16 >= M) return;
    if (n_tile0 * 16 >= N) return;   // uniform per wave

    const bf16_t* Ap = A + (long)z * a_z;
    const bf16_t* Bp = B + (long)z * b_z + (long)e * b_e + (long)g * b_g;

    int hh  = lane >> 4;             // half (0: lanes 0-15, 1: lanes 16-31)
    int loc = lane & 15;

    int aRow = a_rows ? a_rows[m_tile * 16 + loc] : (m_tile * 16 + loc);
    // padding rows (-1) read a shared zeroed dummy row => branch-free inner loop
    const bf16_t* aPtr = (aRow >= 0) ? (Ap + (size_t)aRow * lda) : zrow;

    // per-strip B row bases (each row of B^T is one output column, K-contiguous)
    const bf16_t* bq[NT];
    #pragma unroll
    for (int nt = 0; nt < NT; ++nt)
        bq[nt] = Bp + (size_t)((n_tile0 + nt) * 16 + loc) * K;

    v8f acc[NT];
    #pragma unroll
    for (int nt = 0; nt < NT; ++nt)
        acc[nt] = (v8f){0.f, 0.f, 0.f, 0.f, 0.f, 0.f, 0.f, 0.f};

    // A fragment (16-bit A 16x32): lane(m,half) holds
    //   k = k0 + 16*half + {0..15}  -> one 32B load (matches B)
    auto loadA = [&](int k) -> v16bf {
        return *(const v16bf*)(aPtr + k + 16 * hh);
    };
    // B fragment (32x16): lane(n,half) holds k = k0 + 16*half + {0..15} -> one 32B load
    auto loadB = [&](int nt, int k) -> v16bf {
        return *(const v16bf*)(bq[nt] + k + 16 * hh);
    };

    // 2-stage software pipeline over 64-wide K chunks (K % 64 == 0)
    v16bf af0 = loadA(0), af1;
    v16bf bf0[NT], bf1[NT];
    #pragma unroll
    for (int nt = 0; nt < NT; ++nt) bf0[nt] = loadB(nt, 0);

    for (int k0 = 0; k0 < K; k0 += 64) {
        // issue loads for k0+32, compute k0
        af1 = loadA(k0 + 32);
        #pragma unroll
        for (int nt = 0; nt < NT; ++nt) bf1[nt] = loadB(nt, k0 + 32);
        #pragma unroll
        for (int nt = 0; nt < NT; ++nt)
            acc[nt] = __builtin_amdgcn_wmma_f32_16x16x32_bf16(
                          false, af0, false, bf0[nt], (short)0, acc[nt], false, false);
        // issue loads for k0+64 (uniform branch), compute k0+32
        if (k0 + 64 < K) {
            af0 = loadA(k0 + 64);
            #pragma unroll
            for (int nt = 0; nt < NT; ++nt) bf0[nt] = loadB(nt, k0 + 64);
        }
        #pragma unroll
        for (int nt = 0; nt < NT; ++nt)
            acc[nt] = __builtin_amdgcn_wmma_f32_16x16x32_bf16(
                          false, af1, false, bf1[nt], (short)0, acc[nt], false, false);
    }

    // store: VGPR v of lane -> C[m = v + 8*half][n]
    int crow[8];
    #pragma unroll
    for (int v = 0; v < 8; ++v) {
        int slot = m_tile * 16 + v + 8 * hh;
        crow[v] = c_rows ? c_rows[slot] : slot;
    }
    #pragma unroll
    for (int nt = 0; nt < NT; ++nt) {
        int ncol = (n_tile0 + nt) * 16 + loc;
        #pragma unroll
        for (int v = 0; v < 8; ++v) {
            if (crow[v] >= 0) {
                size_t idx = (size_t)crow[v] * ldc + ncol + (size_t)z * c_z;
                if (Cb) Cb[idx] = (bf16_t)acc[nt][v];
                else    Cf[idx] = acc[nt][v];
            }
        }
    }
}

// ---------------- silu(gate) * up, in-place into gate buffer (bf16) ----------------
__global__ void act_kernel(bf16_t* __restrict__ yg, const bf16_t* __restrict__ yu, long n) {
    long i = (long)blockIdx.x * 256 + threadIdx.x;
    if (i >= n) return;
    float gv = (float)yg[i];
    float s  = gv / (1.f + __expf(-gv));
    yg[i] = (bf16_t)(s * (float)yu[i]);
}

// ---------------- weighted combine over the 6 experts of each token ----------------
__global__ __launch_bounds__(256)
void combine_kernel(const float* __restrict__ yd, const float* __restrict__ weights,
                    float* __restrict__ out) {
    int t = blockIdx.x;
    float w[TOPK];
    #pragma unroll
    for (int k = 0; k < TOPK; ++k) w[k] = weights[t * TOPK + k];
    for (int d = threadIdx.x; d < HH; d += 256) {
        float acc = 0.f;
        #pragma unroll
        for (int k = 0; k < TOPK; ++k)
            acc += w[k] * yd[(size_t)(t * TOPK + k) * HH + d];
        out[(size_t)t * HH + d] = acc;
    }
}

// ---------------- host launcher ----------------
extern "C" void kernel_launch(void* const* d_in, const int* in_sizes, int n_in,
                              void* d_out, int out_size, void* d_ws, size_t ws_size,
                              hipStream_t stream) {
    const float* x   = (const float*)d_in[0];   // [T,H]
    const float* wr  = (const float*)d_in[1];   // [H,E]
    const float* Uig = (const float*)d_in[2];   // [G,H,R1]
    const float* Cg  = (const float*)d_in[3];   // [E,R1,R2]
    const float* Uog = (const float*)d_in[4];   // [G,I,R2]
    const float* Uiu = (const float*)d_in[5];
    const float* Cu  = (const float*)d_in[6];
    const float* Uou = (const float*)d_in[7];
    const float* Uid = (const float*)d_in[8];   // [G,I,R1]
    const float* Cd  = (const float*)d_in[9];
    const float* Uod = (const float*)d_in[10];  // [G,H,R2]
    float* out = (float*)d_out;

    // workspace carve-up (256B aligned)
    char* wp = (char*)d_ws;
    auto carve = [&](size_t bytes) -> void* {
        void* p = (void*)wp;
        wp += (bytes + 255) & ~(size_t)255;
        return p;
    };
    float*  logits   = (float*)carve((size_t)TT * EE * 4);
    int*    selected = (int*)  carve((size_t)PAIRS * 4);
    float*  weights  = (float*)carve((size_t)PAIRS * 4);
    int*    counts   = (int*)  carve(EE * 4);
    int*    cursor   = (int*)  carve(EE * 4);
    int*    pstart   = (int*)  carve(EE * 4);
    int*    tstart   = (int*)  carve(EE * 4);
    int*    tile_exp = (int*)  carve(MAXT * 4);
    int*    sorted_p = (int*)  carve(MAXS * 4);
    int*    rowGH    = (int*)  carve(MAXS * 4);
    bf16_t* zrow     = (bf16_t*)carve((size_t)HH * 2);          // shared zero row (max K)
    // bf16 pre-transformed operands
    bf16_t* xbf   = (bf16_t*)carve((size_t)TT * HH * 2);        // [T,H]
    bf16_t* UigT  = (bf16_t*)carve((size_t)GG * RR * HH * 2);   // [G][R1][H]
    bf16_t* UiuT  = (bf16_t*)carve((size_t)GG * RR * HH * 2);
    bf16_t* CgT   = (bf16_t*)carve((size_t)EE * RR * RR * 2);   // [E][R2][R1]
    bf16_t* CuT   = (bf16_t*)carve((size_t)EE * RR * RR * 2);
    bf16_t* CdT   = (bf16_t*)carve((size_t)EE * RR * RR * 2);
    bf16_t* UogB  = (bf16_t*)carve((size_t)GG * II * RR * 2);   // [G][I][R2] (already N-major)
    bf16_t* UouB  = (bf16_t*)carve((size_t)GG * II * RR * 2);
    bf16_t* UodB  = (bf16_t*)carve((size_t)GG * HH * RR * 2);   // [G][H][R2]
    bf16_t* UidT  = (bf16_t*)carve((size_t)GG * RR * II * 2);   // [G][R1][I]
    // bf16 intermediates
    bf16_t* H1g = (bf16_t*)carve((size_t)GG * TT * RR * 2);     // [G,T,R1]
    bf16_t* H1u = (bf16_t*)carve((size_t)GG * TT * RR * 2);
    bf16_t* h2g = (bf16_t*)carve((size_t)MAXS * RR * 2);        // slot-indexed
    bf16_t* h2u = (bf16_t*)carve((size_t)MAXS * RR * 2);
    bf16_t* h1d = (bf16_t*)carve((size_t)MAXS * RR * 2);
    bf16_t* h2d = (bf16_t*)carve((size_t)MAXS * RR * 2);
    bf16_t* yg  = (bf16_t*)carve((size_t)PAIRS * II * 2);       // pair-indexed
    bf16_t* yu  = (bf16_t*)carve((size_t)PAIRS * II * 2);
    float*  yd  = (float*) carve((size_t)PAIRS * HH * 4);
    (void)ws_size; (void)n_in; (void)in_sizes; (void)out_size;

    auto cvt = [&](const float* src, bf16_t* dst, long n) {
        long n4 = n / 4;
        cvt_bf16_kernel<<<(unsigned)((n4 + 255) / 256), 256, 0, stream>>>(src, dst, n4);
    };
    // grid.y for GEMM: strips = N/64, 4 waves per block
    auto gy = [](int N) { int strips = N / (16 * NT); return (strips + 3) / 4; };

    // 0. zero dummy row + operand pre-transform (bf16 convert / transpose to [N][K])
    zero_int_kernel<<<4, 256, 0, stream>>>((int*)zrow, HH / 2);
    cvt(x,   xbf,  (long)TT * HH);
    cvt(Uog, UogB, (long)GG * II * RR);
    cvt(Uou, UouB, (long)GG * II * RR);
    cvt(Uod, UodB, (long)GG * HH * RR);
    transpose_bf16_kernel<<<dim3(RR / 32, HH / 32, GG), 256, 0, stream>>>(Uig, UigT, HH, RR);
    transpose_bf16_kernel<<<dim3(RR / 32, HH / 32, GG), 256, 0, stream>>>(Uiu, UiuT, HH, RR);
    transpose_bf16_kernel<<<dim3(RR / 32, RR / 32, EE), 256, 0, stream>>>(Cg,  CgT,  RR, RR);
    transpose_bf16_kernel<<<dim3(RR / 32, RR / 32, EE), 256, 0, stream>>>(Cu,  CuT,  RR, RR);
    transpose_bf16_kernel<<<dim3(RR / 32, RR / 32, EE), 256, 0, stream>>>(Cd,  CdT,  RR, RR);
    transpose_bf16_kernel<<<dim3(RR / 32, II / 32, GG), 256, 0, stream>>>(Uid, UidT, II, RR);

    // 1. router (exact fp32) + top-k
    router_kernel<<<TT, 256, 0, stream>>>(x, wr, logits);
    topk_kernel<<<TT, 32, 0, stream>>>(logits, selected, weights);

    // 2. routing plan
    zero_int_kernel<<<1, 256, 0, stream>>>(counts, EE);
    zero_int_kernel<<<1, 256, 0, stream>>>(cursor, EE);
    count_kernel<<<(PAIRS + 255) / 256, 256, 0, stream>>>(selected, counts);
    plan_kernel<<<1, 256, 0, stream>>>(counts, pstart, tstart, tile_exp, sorted_p, rowGH);
    scatter_kernel<<<(PAIRS + 255) / 256, 256, 0, stream>>>(selected, pstart, cursor,
                                                            sorted_p, rowGH);

    // 3. dense in-projections for all 8 groups: H1[g] = X @ U_in[g]   [T,H]x[H,R1]
    wmma_gemm<<<dim3(TT / 16, gy(RR), GG), 128, 0, stream>>>(
        xbf, HH, 0L, nullptr,
        UigT, (long)RR * HH, 0L, 0L,
        H1g, nullptr, RR, (long)TT * RR, nullptr,
        TT, RR, HH, nullptr, zrow);
    wmma_gemm<<<dim3(TT / 16, gy(RR), GG), 128, 0, stream>>>(
        xbf, HH, 0L, nullptr,
        UiuT, (long)RR * HH, 0L, 0L,
        H1u, nullptr, RR, (long)TT * RR, nullptr,
        TT, RR, HH, nullptr, zrow);

    // 4. per-expert core GEMMs: h2[slot] = H1[g,t] @ core[e]   (gather rows via rowGH)
    wmma_gemm<<<dim3(MAXT, gy(RR), 1), 128, 0, stream>>>(
        H1g, RR, 0L, rowGH,
        CgT, 0L, (long)RR * RR, 0L,
        h2g, nullptr, RR, 0L, nullptr,
        MAXS, RR, RR, tile_exp, zrow);
    wmma_gemm<<<dim3(MAXT, gy(RR), 1), 128, 0, stream>>>(
        H1u, RR, 0L, rowGH,
        CuT, 0L, (long)RR * RR, 0L,
        h2u, nullptr, RR, 0L, nullptr,
        MAXS, RR, RR, tile_exp, zrow);

    // 5. out-projections: y[p] = h2[slot] @ U_out[g]^T  (B already [N][K])
    wmma_gemm<<<dim3(MAXT, gy(II), 1), 128, 0, stream>>>(
        h2g, RR, 0L, nullptr,
        UogB, 0L, 0L, (long)II * RR,
        yg, nullptr, II, 0L, sorted_p,
        MAXS, II, RR, tile_exp, zrow);
    wmma_gemm<<<dim3(MAXT, gy(II), 1), 128, 0, stream>>>(
        h2u, RR, 0L, nullptr,
        UouB, 0L, 0L, (long)II * RR,
        yu, nullptr, II, 0L, sorted_p,
        MAXS, II, RR, tile_exp, zrow);

    // 6. inter = silu(gate) * up  (in-place into yg)
    long nact = (long)PAIRS * II;
    act_kernel<<<(unsigned)((nact + 255) / 256), 256, 0, stream>>>(yg, yu, nact);

    // 7. down path: h1d[slot] = inter[p] @ U_in_down[g]  (K = I = 1408)
    wmma_gemm<<<dim3(MAXT, gy(RR), 1), 128, 0, stream>>>(
        yg, II, 0L, sorted_p,
        UidT, 0L, 0L, (long)RR * II,
        h1d, nullptr, RR, 0L, nullptr,
        MAXS, RR, II, tile_exp, zrow);
    //    h2d[slot] = h1d[slot] @ core_down[e]
    wmma_gemm<<<dim3(MAXT, gy(RR), 1), 128, 0, stream>>>(
        h1d, RR, 0L, nullptr,
        CdT, 0L, (long)RR * RR, 0L,
        h2d, nullptr, RR, 0L, nullptr,
        MAXS, RR, RR, tile_exp, zrow);
    //    yd[p] = h2d[slot] @ U_out_down[g]^T   [*,R2]x[R2,H]  (fp32 out)
    wmma_gemm<<<dim3(MAXT, gy(HH), 1), 128, 0, stream>>>(
        h2d, RR, 0L, nullptr,
        UodB, 0L, 0L, (long)HH * RR,
        nullptr, yd, HH, 0L, sorted_p,
        MAXS, HH, RR, tile_exp, zrow);

    // 8. weighted combine per token
    combine_kernel<<<TT, 256, 0, stream>>>(yd, weights, out);
}
